// LinearAttention_27745488732193
// MI455X (gfx1250) — matrix operand order repaired and checked by reading, not compiled, not run on hardware
//
#include <hip/hip_runtime.h>

// ---------------- types ----------------
typedef __attribute__((ext_vector_type(16))) __bf16 v16bf;
typedef __attribute__((ext_vector_type(8)))  __bf16 v8bf;
typedef __attribute__((ext_vector_type(2)))  __bf16 v2bf;
typedef __attribute__((ext_vector_type(8)))  float  v8f;
typedef __attribute__((ext_vector_type(4)))  unsigned short us4;

// ---------------- problem constants ----------------
// S=4096, B=8, D=1024, H=16, HD=64.  Row index r = b*4096 + s, r in [0,32768).
#define MROWS 32768
#define KDIM  1024
#define NDIM  1024

// ---------------- helpers ----------------
__device__ inline unsigned short f32_to_bf16_rne(float f) {
  unsigned int u = __builtin_bit_cast(unsigned int, f);
  u += 0x7FFFu + ((u >> 16) & 1u);
  return (unsigned short)(u >> 16);
}

__device__ inline us4 cvt4_bf16(float4 f) {
  us4 h;
#if __has_builtin(__builtin_amdgcn_cvt_pk_bf16_f32)
  unsigned lo = __builtin_bit_cast(unsigned, __builtin_amdgcn_cvt_pk_bf16_f32(f.x, f.y));
  unsigned hi = __builtin_bit_cast(unsigned, __builtin_amdgcn_cvt_pk_bf16_f32(f.z, f.w));
  h[0] = (unsigned short)lo; h[1] = (unsigned short)(lo >> 16);
  h[2] = (unsigned short)hi; h[3] = (unsigned short)(hi >> 16);
#else
  h[0] = f32_to_bf16_rne(f.x); h[1] = f32_to_bf16_rne(f.y);
  h[2] = f32_to_bf16_rne(f.z); h[3] = f32_to_bf16_rne(f.w);
#endif
  return h;
}

__device__ inline v16bf cat16(v8bf lo, v8bf hi) {
  v16bf r;
#pragma unroll
  for (int i = 0; i < 8; ++i) { r[i] = lo[i]; r[i + 8] = hi[i]; }
  return r;
}

__device__ inline float wredsum(float x) {
#pragma unroll
  for (int i = 16; i > 0; i >>= 1) x += __shfl_xor(x, i, 32);
  return x;
}
__device__ inline float wredmax(float x) {
#pragma unroll
  for (int i = 16; i > 0; i >>= 1) x = fmaxf(x, __shfl_xor(x, i, 32));
  return x;
}

// ---------------- WMMA GEMM:  Out[r][n] = sum_k Xrow(r)[k] * W[n][k] + bias[n]
enum { AM_QKV_F32 = 0, AM_LIN_BF16 = 1 };
enum { OM_LIN_F32 = 0, OM_SCATTER_F32 = 1 };

template <int AM, int OM>
__global__ __launch_bounds__(256) void wmma_gemm_1024(
    const void* __restrict__ Xv, const float* __restrict__ W,
    const float* __restrict__ bias, float* __restrict__ Out) {
  constexpr int BM = 128, BN = 128, BK = 32, BKP = 40;  // 80B row stride: 16B-aligned, bank-spread
  __shared__ unsigned short As[2][BM * BKP];
  __shared__ unsigned short Bs[2][BN * BKP];

  const int tid  = threadIdx.x;
  const int lane = tid & 31;
  const int wv   = tid >> 5;           // 8 waves: 4 (M) x 2 (N)
  const int wm   = (wv & 3) * 32;      // wave tile: 32 rows x 64 cols
  const int wn   = (wv >> 2) * 64;
  const int rowBlk = blockIdx.y * BM;
  const int colBlk = blockIdx.x * BN;

  v8f acc[2][4] = {};

  // staging: 2 threads per row, 16 k-elements each (covers 128 rows x 32 k)
  const int srow = tid >> 1, sk0 = (tid & 1) * 16;

  size_t aRowOff;
  if constexpr (AM == AM_QKV_F32) {
    const int gr = rowBlk + srow;                // r = b*4096 + s
    aRowOff = (size_t)(gr & 4095) * 8192 + (size_t)(gr >> 12) * 1024;  // x[s][b][*]
  } else {
    aRowOff = (size_t)(rowBlk + srow) * KDIM;
  }
  const size_t bRowOff = (size_t)(colBlk + srow) * KDIM;  // rows of W (NT gemm)

  const float* Xf = (const float*)Xv;
  float4 ra[4], rb[4];

  auto loadA = [&](int kt) {
#pragma unroll
    for (int i = 0; i < 4; ++i) ra[i] = *(const float4*)(Xf + aRowOff + kt + sk0 + 4 * i);
  };
  auto loadB = [&](int kt) {
#pragma unroll
    for (int i = 0; i < 4; ++i) rb[i] = *(const float4*)(W + bRowOff + kt + sk0 + 4 * i);
  };
  auto storeA = [&](int buf) {
#pragma unroll
    for (int i = 0; i < 4; ++i) *(us4*)&As[buf][srow * BKP + sk0 + 4 * i] = cvt4_bf16(ra[i]);
  };
  auto storeB = [&](int buf) {
#pragma unroll
    for (int i = 0; i < 4; ++i) *(us4*)&Bs[buf][srow * BKP + sk0 + 4 * i] = cvt4_bf16(rb[i]);
  };
  // bf16 A rows need no conversion: DMA straight to LDS (ASYNCcnt-tracked).
  // ISA: INST_OFFSET is added to BOTH the global and the LDS address.
  auto asyncA = [&](int kt, int buf) {
    const unsigned long long base = (unsigned long long)(uintptr_t)Xv;
    const unsigned goff = (unsigned)((aRowOff + (size_t)kt + sk0) * 2u);          // bytes
    const unsigned loff = (unsigned)(uintptr_t)&As[buf][srow * BKP + sk0];        // low32 = LDS addr
    asm volatile("global_load_async_to_lds_b128 %0, %1, %2"
                 :: "v"(loff), "v"(goff), "s"(base) : "memory");
    asm volatile("global_load_async_to_lds_b128 %0, %1, %2 offset:16"
                 :: "v"(loff), "v"(goff), "s"(base) : "memory");
  };

  // ---- prologue: fill buffer 0 ----
  if constexpr (AM == AM_LIN_BF16) {
    asyncA(0, 0);
  } else {
    loadA(0); storeA(0);
  }
  loadB(0); storeB(0);
  if constexpr (AM == AM_LIN_BF16)
    asm volatile("s_wait_asynccnt 0x0" ::: "memory");
  __syncthreads();

  // ---- main loop: double-buffered ----
  for (int kt = 0; kt < KDIM; kt += BK) {
    const int buf = (kt >> 5) & 1;
    const bool hasNext = (kt + BK) < KDIM;
    if (hasNext) {
      if constexpr (AM == AM_LIN_BF16) asyncA(kt + BK, buf ^ 1);
      else                             loadA(kt + BK);
      loadB(kt + BK);
    }

    // fragments per CDNA5 16-bit WMMA VGPR layouts
    const int fr  = lane & 15;
    const int akb = (lane < 16) ? 0 : 8;    // A: lane<16 -> K 0..7 & 16..23 ; else 8..15 & 24..31
    const int bkb = (lane < 16) ? 0 : 8;    // B: lane<16 -> K 0..7 & 16..23 ; else 8..15 & 24..31
    v16bf a[2], b[4];
#pragma unroll
    for (int mi = 0; mi < 2; ++mi) {
      const unsigned short* p = &As[buf][(wm + mi * 16 + fr) * BKP + akb];
      a[mi] = cat16(*(const v8bf*)p, *(const v8bf*)(p + 16));
    }
#pragma unroll
    for (int ni = 0; ni < 4; ++ni) {
      const unsigned short* p = &Bs[buf][(wn + ni * 16 + fr) * BKP + bkb];
      b[ni] = cat16(*(const v8bf*)p, *(const v8bf*)(p + 16));
    }
#pragma unroll
    for (int mi = 0; mi < 2; ++mi)
#pragma unroll
      for (int ni = 0; ni < 4; ++ni)
        acc[mi][ni] = __builtin_amdgcn_wmma_f32_16x16x32_bf16(
            false, a[mi], false, b[ni], (short)0, acc[mi][ni], false, false);

    if (hasNext) {
      if constexpr (AM != AM_LIN_BF16) storeA(buf ^ 1);
      storeB(buf ^ 1);
      if constexpr (AM == AM_LIN_BF16)
        asm volatile("s_wait_asynccnt 0x0" ::: "memory");
    }
    __syncthreads();
  }

  // ---- epilogue: bias + store (C/D layout: VGPR v -> row v, +8 for lanes 16..31) ----
#pragma unroll
  for (int ni = 0; ni < 4; ++ni) {
    const int col = colBlk + wn + ni * 16 + (lane & 15);
    const float bv = bias[col];
#pragma unroll
    for (int mi = 0; mi < 2; ++mi) {
      const int row0 = rowBlk + wm + mi * 16 + ((lane < 16) ? 0 : 8);
#pragma unroll
      for (int v = 0; v < 8; ++v) {
        const float val = acc[mi][ni][v] + bv;
        const int row = row0 + v;
        if constexpr (OM == OM_LIN_F32) {
          Out[(size_t)row * NDIM + col] = val;
        } else {
          const int s = row & 4095, bb = row >> 12;   // r = b*4096 + s
          Out[(size_t)(s & 7) * 4194304 + (size_t)(bb * 512 + (s >> 3)) * 1024 + col] = val;
        }
      }
    }
  }
}

// ---------------- linear-attention core + fused softmax (bandwidth-bound) ----
// One wave per (b, sblk, h) group; q,k,v are 8x64.
// A = softmax(q) softmax(k)^T (8x8);  out = (A v) / rowsum(A)
__global__ __launch_bounds__(256) void linattn_core(
    const float* __restrict__ Yq, const float* __restrict__ Yk,
    const float* __restrict__ Yv, unsigned short* __restrict__ o3) {
  const int lane = threadIdx.x & 31;
  const int gid  = blockIdx.x * 8 + (threadIdx.x >> 5);
  const int h    = gid & 15;
  const int r8   = gid >> 4;                                  // b*512 + sblk
  const size_t base0 = (size_t)r8 * 8 * 1024 + (size_t)h * 64;

  float qa[8], qb[8], ka[8], kb2[8], va[8], vb[8];
#pragma unroll
  for (int n = 0; n < 8; ++n) {
    const size_t off = base0 + (size_t)n * 1024;
    qa[n] = Yq[off + lane];  qb[n] = Yq[off + lane + 32];
    ka[n] = Yk[off + lane];  kb2[n] = Yk[off + lane + 32];
    va[n] = Yv[off + lane];  vb[n] = Yv[off + lane + 32];
  }

#pragma unroll
  for (int n = 0; n < 8; ++n) {
    float m = wredmax(fmaxf(qa[n], qb[n]));
    float e0 = __expf(qa[n] - m), e1 = __expf(qb[n] - m);
    float inv = 1.0f / wredsum(e0 + e1);
    qa[n] = e0 * inv; qb[n] = e1 * inv;

    m = wredmax(fmaxf(ka[n], kb2[n]));
    e0 = __expf(ka[n] - m); e1 = __expf(kb2[n] - m);
    inv = 1.0f / wredsum(e0 + e1);
    ka[n] = e0 * inv; kb2[n] = e1 * inv;
  }

  float oa[8] = {}, ob[8] = {}, Z[8] = {};
#pragma unroll
  for (int m = 0; m < 8; ++m) {
    float am[8];
#pragma unroll
    for (int n = 0; n < 8; ++n)
      am[n] = wredsum(qa[n] * ka[m] + qb[n] * kb2[m]);
#pragma unroll
    for (int n = 0; n < 8; ++n) {
      oa[n] += am[n] * va[m];
      ob[n] += am[n] * vb[m];
      Z[n]  += am[n];
    }
  }

#pragma unroll
  for (int n = 0; n < 8; ++n) {
    const float inv = 1.0f / Z[n];
    const size_t off = base0 + (size_t)n * 1024;
    o3[off + lane]      = f32_to_bf16_rne(oa[n] * inv);
    o3[off + lane + 32] = f32_to_bf16_rne(ob[n] * inv);
  }
}

// ---------------- launcher ----------------
extern "C" void kernel_launch(void* const* d_in, const int* in_sizes, int n_in,
                              void* d_out, int out_size, void* d_ws, size_t ws_size,
                              hipStream_t stream) {
  (void)in_sizes; (void)n_in; (void)out_size; (void)ws_size;
  const float* q  = (const float*)d_in[0];
  const float* k  = (const float*)d_in[1];
  const float* v  = (const float*)d_in[2];
  const float* Wq = (const float*)d_in[3];  const float* bq = (const float*)d_in[4];
  const float* Wk = (const float*)d_in[5];  const float* bk = (const float*)d_in[6];
  const float* Wv = (const float*)d_in[7];  const float* bv = (const float*)d_in[8];
  const float* Wo = (const float*)d_in[9];  const float* bo = (const float*)d_in[10];
  float* out = (float*)d_out;

  const size_t SZ = (size_t)MROWS * NDIM;
  float* Yq = (float*)d_ws;
  float* Yk = Yq + SZ;
  float* Yv = Yk + SZ;
  unsigned short* o3 = (unsigned short*)(Yv + SZ);   // bf16

  dim3 grid(NDIM / 128, MROWS / 128);                // 8 x 256
  dim3 blk(256);

  wmma_gemm_1024<AM_QKV_F32, OM_LIN_F32><<<grid, blk, 0, stream>>>(q, Wq, bq, Yq);
  wmma_gemm_1024<AM_QKV_F32, OM_LIN_F32><<<grid, blk, 0, stream>>>(k, Wk, bk, Yk);
  wmma_gemm_1024<AM_QKV_F32, OM_LIN_F32><<<grid, blk, 0, stream>>>(v, Wv, bv, Yv);

  linattn_core<<<8192, 256, 0, stream>>>(Yq, Yk, Yv, o3);

  wmma_gemm_1024<AM_LIN_BF16, OM_SCATTER_F32><<<grid, blk, 0, stream>>>(o3, Wo, bo, out);
}
